// BahdanauAttention_69226282877122
// MI455X (gfx1250) — matrix operand order repaired and checked by reading, not compiled, not run on hardware
//
#include <hip/hip_runtime.h>
#include <hip/hip_bf16.h>
#include <math.h>

// ---- problem constants (from reference) ----
#define B_   256
#define L_   225
#define D_   1280
#define H_   256
#define U_   256

// ---- tiling ----
#define KC      128              // K-chunk staged in LDS
#define NCHUNK  (D_ / KC)        // 10
#define KS      (KC / 32)        // 4 WMMA K-steps per chunk
#define MT      15               // M tiles of 16 -> 240 padded rows
#define MROWS   (MT * 16)        // 240
#define THREADS 512              // 16 waves (wave32)

// ---- dynamic LDS layout (bytes) ----
#define LDSB_OFF   0
#define LDSB_BYTES (U_ * KC * 2)        // W1^T chunk, bf16: 65536
#define LDSA_OFF   (LDSB_OFF + LDSB_BYTES)
#define LDSA_BYTES (MROWS * KC * 2)     // features chunk, bf16: 61440
#define LOG_OFF    (LDSA_OFF + LDSA_BYTES)
#define LOG_BYTES  (MROWS * 4)          // logits / attn
#define RED_OFF    (LOG_OFF + LOG_BYTES)
#define RED_BYTES  (THREADS * 4)        // softmax reduction scratch
#define SMEM_BYTES (RED_OFF + RED_BYTES) // ~129.9 KB < 320 KB WGP LDS

typedef __attribute__((ext_vector_type(8)))  __bf16 v8bf;
typedef __attribute__((ext_vector_type(16))) __bf16 v16bf;
typedef __attribute__((ext_vector_type(8)))  float  v8f;

// Load a 16-element bf16 fragment from two 16B LDS slots (-> 2x ds_load_b128)
__device__ __forceinline__ v16bf frag16(const __bf16* p0, const __bf16* p1) {
    const v8bf lo = *(const v8bf*)p0;
    const v8bf hi = *(const v8bf*)p1;
    v16bf r;
#pragma unroll
    for (int j = 0; j < 8; ++j) { r[j] = lo[j]; r[j + 8] = hi[j]; }
    return r;
}

// tanh(x) = 1 - 2/(exp(2x)+1): one v_exp_f32 TRANS op instead of OCML poly.
__device__ __forceinline__ float fast_tanh(float x) {
    return 1.0f - 2.0f / (__expf(2.0f * x) + 1.0f);
}

// ph[b,u] = b1[u] + b2[u] + hidden[b,:] . W2[:,u]   (tiny: 33 MFLOP)
__global__ void bahdanau_projh(const float* __restrict__ hidden,
                               const float* __restrict__ W2,
                               const float* __restrict__ b1,
                               const float* __restrict__ b2,
                               float* __restrict__ ph) {
    const int b = blockIdx.x;
    const int u = threadIdx.x;
    float s = b1[u] + b2[u];
    const float* hrow = hidden + b * H_;
#pragma unroll 4
    for (int h = 0; h < H_; ++h) s += hrow[h] * W2[h * U_ + u];
    ph[b * U_ + u] = s;
}

__global__ __launch_bounds__(THREADS, 1)
void bahdanau_main(const float* __restrict__ F,    // [B, L, D]
                   const float* __restrict__ W1,   // [D, U]
                   const float* __restrict__ Vv,   // [U]
                   const float* __restrict__ ph,   // [B, U]
                   float* __restrict__ out)        // [B, D]
{
    extern __shared__ char smem[];
    __bf16* ldsB    = (__bf16*)(smem + LDSB_OFF);  // [U][KC]   W1^T chunk
    __bf16* ldsA    = (__bf16*)(smem + LDSA_OFF);  // [240][KC] features chunk
    float*  logitsS = (float*)(smem + LOG_OFF);    // [240]
    float*  red     = (float*)(smem + RED_OFF);    // [512]

    const int b    = blockIdx.x;
    const int tid  = threadIdx.x;
    const int wave = tid >> 5;        // n-tile id 0..15
    const int lane = tid & 31;
    const int half = lane >> 4;       // 0: lanes 0-15, 1: lanes 16-31
    const int l16  = lane & 15;

    if (tid < MROWS) logitsS[tid] = 0.0f;

    // Per-wave f32 accumulators: 15 M-tiles x (16x16)/32 lanes = 120 VGPRs.
    v8f zero;
#pragma unroll
    for (int j = 0; j < 8; ++j) zero[j] = 0.0f;
    v8f acc[MT];
#pragma unroll
    for (int m = 0; m < MT; ++m) acc[m] = zero;

    const float* Fb = F + (size_t)b * L_ * D_;

    // Per-lane LDS fragment bases (ISA 7.12.2 bf16 layouts):
    //  A (16x32): row M=l16, K in {half*8+j, 16+half*8+j}
    //  B (32x16): col N=wave*16+l16, K in {half*8+j, 16+half*8+j}
    const __bf16* aBase = ldsA + l16 * KC + half * 8;
    const __bf16* bBase = ldsB + (wave * 16 + l16) * KC + half * 8;

    for (int c = 0; c < NCHUNK; ++c) {
        const int k0g = c * KC;
        __syncthreads();
        // Stage W1^T chunk as bf16: ldsB[n][k]. Global read coalesced over n.
        for (int idx = tid; idx < U_ * KC; idx += THREADS) {
            const int k = idx / U_;
            const int n = idx % U_;
            ldsB[n * KC + k] = (__bf16)W1[(size_t)(k0g + k) * U_ + n];
        }
        // Stage features chunk as bf16: ldsA[row][k], rows >= L zero-padded.
        for (int idx = tid; idx < MROWS * KC; idx += THREADS) {
            const int r  = idx / KC;
            const int kk = idx % KC;
            const float f = (r < L_) ? Fb[(size_t)r * D_ + (k0g + kk)] : 0.0f;
            ldsA[r * KC + kk] = (__bf16)f;
        }
        __syncthreads();

        // Hoist all 4 B fragments for this chunk (32 VGPRs, loads in flight).
        v16bf bf[KS];
#pragma unroll
        for (int ks = 0; ks < KS; ++ks)
            bf[ks] = frag16(bBase + ks * 32, bBase + ks * 32 + 16);

        // Software-pipelined A fragments: load m+1 before WMMA m so the
        // waitcnt pass can emit s_wait_dscnt 2 instead of 0 (LDS latency
        // overlaps WMMA execution).
        v16bf a_cur = frag16(aBase, aBase + 16);
#pragma unroll
        for (int ks = 0; ks < KS; ++ks) {
            const int k0 = ks * 32;
#pragma unroll
            for (int m = 0; m < MT; ++m) {
                v16bf a_next;
                if (m < MT - 1) {
                    const __bf16* ap = aBase + (m + 1) * 16 * KC + k0;
                    a_next = frag16(ap, ap + 16);
                } else if (ks < KS - 1) {
                    const __bf16* ap = aBase + (k0 + 32);
                    a_next = frag16(ap, ap + 16);
                } else {
                    a_next = a_cur;  // drained; dummy
                }
                acc[m] = __builtin_amdgcn_wmma_f32_16x16x32_bf16(
                    false, a_cur, false, bf[ks], (short)0, acc[m], false, false);
                a_cur = a_next;
            }
        }
    }
    __syncthreads();

    // Epilogue: score = tanh(acc + ph), logits += score * V[u].
    // C/D layout: VGPR r, lanes 0-15 -> M=r, lanes 16-31 -> M=r+8; N = l16.
    {
        const int   ncol = wave * 16 + l16;
        const float phv  = ph[b * U_ + ncol];
        const float vv   = Vv[ncol];
#pragma unroll
        for (int m = 0; m < MT; ++m) {
#pragma unroll
            for (int r = 0; r < 8; ++r) {
                float s = fast_tanh(acc[m][r] + phv) * vv;
                s += __shfl_xor(s, 1, 32);   // reduce over the 16 N-lanes
                s += __shfl_xor(s, 2, 32);   // (masks < 16 stay within half)
                s += __shfl_xor(s, 4, 32);
                s += __shfl_xor(s, 8, 32);
                if (l16 == 0)
                    atomicAdd(&logitsS[m * 16 + half * 8 + r], s);  // ds_add_f32
            }
        }
    }
    __syncthreads();

    // Softmax over l = 0..224 (bV is constant -> cancels in softmax).
    const float lv = (tid < L_) ? logitsS[tid] : -1.0e30f;
    red[tid] = lv;
    __syncthreads();
    for (int s = THREADS / 2; s > 0; s >>= 1) {
        if (tid < s) red[tid] = fmaxf(red[tid], red[tid + s]);
        __syncthreads();
    }
    const float mx = red[0];
    __syncthreads();
    const float ev = (tid < L_) ? __expf(lv - mx) : 0.0f;
    red[tid] = ev;
    __syncthreads();
    for (int s = THREADS / 2; s > 0; s >>= 1) {
        if (tid < s) red[tid] += red[tid + s];
        __syncthreads();
    }
    const float inv = 1.0f / red[0];
    if (tid < L_) logitsS[tid] = ev * inv;   // attn weights
    __syncthreads();

    // context[b, d] = sum_l attn[l] * features[b, l, d]  (features L2-hot)
    for (int d = tid; d < D_; d += THREADS) {
        float cacc = 0.0f;
#pragma unroll 5
        for (int l = 0; l < L_; ++l)
            cacc += logitsS[l] * Fb[(size_t)l * D_ + d];
        out[(size_t)b * D_ + d] = cacc;
    }
}

extern "C" void kernel_launch(void* const* d_in, const int* in_sizes, int n_in,
                              void* d_out, int out_size, void* d_ws, size_t ws_size,
                              hipStream_t stream) {
    const float* features = (const float*)d_in[0];  // [B, L, D]
    const float* hidden   = (const float*)d_in[1];  // [B, H]
    const float* W1       = (const float*)d_in[2];  // [D, U]
    const float* b1       = (const float*)d_in[3];  // [U]
    const float* W2       = (const float*)d_in[4];  // [H, U]
    const float* b2       = (const float*)d_in[5];  // [U]
    const float* V        = (const float*)d_in[6];  // [U, 1]
    // d_in[7] = bV: additive constant, cancels in softmax.
    float* out = (float*)d_out;                     // [B, D]
    float* ph  = (float*)d_ws;                      // [B, U] scratch (256 KB)

    bahdanau_projh<<<B_, U_, 0, stream>>>(hidden, W2, b1, b2, ph);

    (void)hipFuncSetAttribute((const void*)bahdanau_main,
                              hipFuncAttributeMaxDynamicSharedMemorySize,
                              SMEM_BYTES);
    bahdanau_main<<<B_, THREADS, SMEM_BYTES, stream>>>(features, W1, V, ph, out);
}
